// LSTM_1layer_11708080849330
// MI455X (gfx1250) — matrix operand order repaired and checked by reading, not compiled, not run on hardware
//
#include <hip/hip_runtime.h>
#include <math.h>

// ---------------------------------------------------------------------------
// LSTM on MI455X (gfx1250, wave32, WMMA).
// Persistent-kernel recurrence: 16 WGs x 256 threads, grid barrier per step.
// Per step: [128 x 768(D+H)] x [768 x 2048(4 gates)] via v_wmma_f32_16x16x32_f16,
// B matrix resident in LDS (192KB of the 320KB WGP LDS), c state in registers.
// ---------------------------------------------------------------------------

typedef _Float16 h16_t;
typedef _Float16 v16h __attribute__((ext_vector_type(16)));
typedef _Float16 v8h  __attribute__((ext_vector_type(8)));
typedef float    v8f  __attribute__((ext_vector_type(8)));
typedef float    v4f  __attribute__((ext_vector_type(4)));

#define Bsz  128
#define Tsz  1024
#define Dsz  256
#define Hsz  512
#define Csz  1000
#define NKT  24          // (D+H)/32 K-tiles
#define XKT  8           // D/32
#define HKT  16          // H/32
#define NWG  16          // workgroups in persistent grid
#define THREADS 256
#define NTW  63          // ceil(1008/16) column tiles for head (1000 padded to 1008)

// Workspace layout (bytes):
//   [0, 256)                      : grid-barrier {count, sense}
//   [256, 256+3145728)            : packed gate weights  [128 ntile][24 kt][32 lane][16 h]
//   [+3145728, +1032192)          : packed head weights  [63 ntile][16 kt][32 lane][16 h]
//   [..., +262144)                : h double buffer, f16 [2][128][512]
#define WS_WPACK_OFF   256
#define WS_WP_OFF      (WS_WPACK_OFF + 128*NKT*32*16*2)
#define WS_HBUF_OFF    (WS_WP_OFF   + NTW*HKT*32*16*2)

__device__ __forceinline__ float sigmoidf_fast(float v) {
    return 1.0f / (1.0f + __expf(-v));
}

// ---------------------------------------------------------------------------
// Init: zero h(t=0) buffer and barrier state (re-run every call: deterministic)
// ---------------------------------------------------------------------------
__global__ void init_state(h16_t* __restrict__ hbuf, unsigned int* __restrict__ bar) {
    int idx = blockIdx.x * blockDim.x + threadIdx.x;
    if (idx < Bsz * Hsz) hbuf[idx] = (h16_t)0.0f;
    if (idx < 2) bar[idx] = 0u;
}

// ---------------------------------------------------------------------------
// Pack the 4 gate weight pairs (Wx: DxH, Wh: HxH, f32 row-major) into the
// WMMA B-fragment layout in f16. Combined K = 768 (x rows 0..255, h rows
// 256..767); combined N = 2048 (gate g occupies columns [g*512,(g+1)*512)).
// Element [ntile][kt][lane][j]: n = ntile*16 + lane%16,
//                              k = kt*32 + (lane<16?0:8) + j + (j<8?0:8).
// ---------------------------------------------------------------------------
__global__ void pack_gates(const float* __restrict__ Wgx, const float* __restrict__ Wgh,
                           const float* __restrict__ Wix, const float* __restrict__ Wih,
                           const float* __restrict__ Wfx, const float* __restrict__ Wfh,
                           const float* __restrict__ Wox, const float* __restrict__ Woh,
                           h16_t* __restrict__ Wpack) {
    int idx = blockIdx.x * blockDim.x + threadIdx.x;   // [0, 128*24*32)
    if (idx >= 128 * NKT * 32) return;
    int lane  = idx & 31;
    int kt    = (idx >> 5) % NKT;
    int ntile = idx / (32 * NKT);
    int n     = ntile * 16 + (lane & 15);              // global column 0..2047
    int g     = n >> 9;                                // gate index 0..3
    int ngc   = n & 511;                               // column within gate
    int kbase = kt * 32 + ((lane & 16) ? 8 : 0);
    const float* Wx; const float* Wh;
    switch (g) {
        case 0:  Wx = Wgx; Wh = Wgh; break;
        case 1:  Wx = Wix; Wh = Wih; break;
        case 2:  Wx = Wfx; Wh = Wfh; break;
        default: Wx = Wox; Wh = Woh; break;
    }
    h16_t* dst = Wpack + (size_t)idx * 16;
    #pragma unroll
    for (int j = 0; j < 16; ++j) {
        int k = kbase + j + ((j < 8) ? 0 : 8);
        float v = (k < Dsz) ? Wx[(size_t)k * Hsz + ngc]
                            : Wh[(size_t)(k - Dsz) * Hsz + ngc];
        dst[j] = (h16_t)v;
    }
}

// Pack head weights Wph [512 x 1000] -> f16 B-fragments, N padded to 1008.
__global__ void pack_wp(const float* __restrict__ Wph, h16_t* __restrict__ WpPack) {
    int idx = blockIdx.x * blockDim.x + threadIdx.x;   // [0, 63*16*32)
    if (idx >= NTW * HKT * 32) return;
    int lane  = idx & 31;
    int kt    = (idx >> 5) % HKT;
    int ntile = idx / (32 * HKT);
    int n     = ntile * 16 + (lane & 15);              // 0..1007
    int kbase = kt * 32 + ((lane & 16) ? 8 : 0);
    h16_t* dst = WpPack + (size_t)idx * 16;
    #pragma unroll
    for (int j = 0; j < 16; ++j) {
        int k = kbase + j + ((j < 8) ? 0 : 8);
        float v = (n < Csz) ? Wph[(size_t)k * Csz + n] : 0.0f;
        dst[j] = (h16_t)v;
    }
}

// ---------------------------------------------------------------------------
// A-fragment helpers (16x32 f16 tile per the CDNA5 VGPR layout).
// Lane<16 holds K {base..base+7, base+16..base+23} with base=0; lanes>=16 base=8.
// ---------------------------------------------------------------------------
__device__ __forceinline__ v16h load_a_f16(const h16_t* __restrict__ p) {
    v8h lo = *(const v8h*)(p);        // K base..base+7
    v8h hi = *(const v8h*)(p + 16);   // K base+16..base+23
    return __builtin_shufflevector(lo, hi, 0, 1, 2, 3, 4, 5, 6, 7,
                                           8, 9, 10, 11, 12, 13, 14, 15);
}

__device__ __forceinline__ v16h load_a_f32cvt(const float* __restrict__ p) {
    v4f p0 = *(const v4f*)(p);
    v4f p1 = *(const v4f*)(p + 4);
    v4f p2 = *(const v4f*)(p + 16);
    v4f p3 = *(const v4f*)(p + 20);
    v16h a;
    a[0]  = (h16_t)p0[0]; a[1]  = (h16_t)p0[1]; a[2]  = (h16_t)p0[2]; a[3]  = (h16_t)p0[3];
    a[4]  = (h16_t)p1[0]; a[5]  = (h16_t)p1[1]; a[6]  = (h16_t)p1[2]; a[7]  = (h16_t)p1[3];
    a[8]  = (h16_t)p2[0]; a[9]  = (h16_t)p2[1]; a[10] = (h16_t)p2[2]; a[11] = (h16_t)p2[3];
    a[12] = (h16_t)p3[0]; a[13] = (h16_t)p3[1]; a[14] = (h16_t)p3[2]; a[15] = (h16_t)p3[3];
    return a;
}

// ---------------------------------------------------------------------------
// Persistent recurrence kernel. Grid = 16 WGs x 256 threads.
// WG wg owns, for each gate g, columns [wg*32, wg*32+32) of that gate's 512.
// Wave w owns batch rows [w*16, w*16+16). 8 acc tiles/wave (t8 = g*2 + ct).
// ---------------------------------------------------------------------------
__global__ __launch_bounds__(THREADS) void lstm_persist(
    const float* __restrict__ x,                       // [B][T][D] f32
    const float* __restrict__ bg, const float* __restrict__ bi,
    const float* __restrict__ bf, const float* __restrict__ bo,
    const h16_t* __restrict__ Wpack,                   // packed gate weights
    h16_t* __restrict__ hbuf,                          // [2][B*H] f16
    unsigned int* __restrict__ bar)                    // {count, sense}
{
    extern __shared__ h16_t ldsB[];                    // [8 t8][24 kt][32 lane][16]
    const int tid  = threadIdx.x;
    const int wg   = blockIdx.x;                       // 0..15
    const int wave = tid >> 5;                         // 0..7 (row tile)
    const int lane = tid & 31;
    const int l16  = lane & 15;
    const int hi8  = (lane >> 4) & 1;                  // 0: lanes 0-15, 1: lanes 16-31

    // ---- Stage this WG's B slice (8 col-tiles x full K) into LDS, once. ----
    for (int ic = tid; ic < 8 * NKT * 32 * 2; ic += THREADS) {
        int hsel  = ic & 1;                            // which 8-half chunk
        int p     = ic >> 1;                           // (t8, kt, lane) flat
        int plane = p & 31;
        int pkt   = (p >> 5) % NKT;
        int pt8   = p / (32 * NKT);
        int ntile = (pt8 >> 1) * 32 + wg * 2 + (pt8 & 1);   // global column tile
        *(v8h*)&ldsB[((pt8 * NKT + pkt) * 32 + plane) * 16 + hsel * 8] =
            *(const v8h*)&Wpack[(((size_t)ntile * NKT + pkt) * 32 + plane) * 16 + hsel * 8];
    }
    __syncthreads();

    // ---- Per-lane constants: bias per acc tile (n fixed per lane/tile). ----
    const int ng0 = wg * 32 + l16;                     // ct = 0 column
    const int ng1 = ng0 + 16;                          // ct = 1 column
    float biasv[8];
    biasv[0] = bg[ng0]; biasv[1] = bg[ng1];
    biasv[2] = bi[ng0]; biasv[3] = bi[ng1];
    biasv[4] = bf[ng0]; biasv[5] = bf[ng1];
    biasv[6] = bo[ng0]; biasv[7] = bo[ng1];

    const int brow = wave * 16 + l16;                  // A-row (batch) this lane feeds

    float cst[2][8];                                   // cell state, registers
    #pragma unroll
    for (int ct = 0; ct < 2; ++ct)
        #pragma unroll
        for (int j = 0; j < 8; ++j) cst[ct][j] = 0.0f;

    #pragma unroll 1
    for (int t = 0; t < Tsz; ++t) {
        const h16_t* hprev = hbuf + (size_t)(t & 1) * (Bsz * Hsz);
        h16_t*       hnext = hbuf + (size_t)((t & 1) ^ 1) * (Bsz * Hsz);
        const float* xrow  = x + ((size_t)brow * Tsz + t) * Dsz;

        if (t + 1 < Tsz)                               // CDNA5 global_prefetch_b8
            __builtin_prefetch(xrow + Dsz, 0, 1);

        v8f acc[8];
        v8f zacc = {};
        #pragma unroll
        for (int i = 0; i < 8; ++i) acc[i] = zacc;

        // ---- K-part 1: x_t (f32 -> f16 on the fly), kt 0..7 ----
        #pragma unroll 1
        for (int kt = 0; kt < XKT; ++kt) {
            v16h a = load_a_f32cvt(xrow + kt * 32 + hi8 * 8);
            #pragma unroll
            for (int t8 = 0; t8 < 8; ++t8) {
                v16h b = *(const v16h*)&ldsB[((t8 * NKT + kt) * 32 + lane) * 16];
                acc[t8] = __builtin_amdgcn_wmma_f32_16x16x32_f16(
                    false, a, false, b, (short)0, acc[t8], false, false);
            }
        }
        // ---- K-part 2: h_{t-1} (f16), kt 8..23 ----
        #pragma unroll 1
        for (int kh = 0; kh < HKT; ++kh) {
            v16h a = load_a_f16(hprev + (size_t)brow * Hsz + kh * 32 + hi8 * 8);
            #pragma unroll
            for (int t8 = 0; t8 < 8; ++t8) {
                v16h b = *(const v16h*)&ldsB[((t8 * NKT + (XKT + kh)) * 32 + lane) * 16];
                acc[t8] = __builtin_amdgcn_wmma_f32_16x16x32_f16(
                    false, a, false, b, (short)0, acc[t8], false, false);
            }
        }

        // ---- Elementwise LSTM cell; c in registers, h -> global f16. ----
        #pragma unroll
        for (int ct = 0; ct < 2; ++ct) {
            const int nh = wg * 32 + ct * 16 + l16;    // hidden column
            #pragma unroll
            for (int j = 0; j < 8; ++j) {
                float gv = tanhf(acc[0 + ct][j] + biasv[0 + ct]);
                float iv = sigmoidf_fast(acc[2 + ct][j] + biasv[2 + ct]);
                float fv = sigmoidf_fast(acc[4 + ct][j] + biasv[4 + ct]);
                float ov = sigmoidf_fast(acc[6 + ct][j] + biasv[6 + ct]);
                float cv = gv * iv + cst[ct][j] * fv;
                cst[ct][j] = cv;
                float hv = tanhf(cv) * ov;
                int b = wave * 16 + j + hi8 * 8;       // C-tile M mapping
                hnext[(size_t)b * Hsz + nh] = (h16_t)hv;
            }
        }

        // ---- Grid-wide step barrier (release stores, arrive, acquire). ----
        __threadfence();
        __syncthreads();
        if (tid == 0) {
            unsigned arrived = atomicAdd(&bar[0], 1u);
            if (arrived == NWG - 1) {
                bar[0] = 0u;
                __threadfence();
                atomicExch(&bar[1], (unsigned)(t + 1));
            } else {
                while (atomicAdd(&bar[1], 0u) < (unsigned)(t + 1)) {
                    __builtin_amdgcn_s_sleep(2);
                }
            }
        }
        __syncthreads();
        __threadfence();                               // acquire h from other WGs
    }
}

// ---------------------------------------------------------------------------
// Head GEMM: out[128 x 1000] = h_final[128 x 512] @ Wph + bp.
// 63 WGs (one 16-col tile each) x 8 waves (one 16-row tile each).
// ---------------------------------------------------------------------------
__global__ __launch_bounds__(THREADS) void head_gemm(
    const h16_t* __restrict__ hfin,                    // hbuf[0]: [B][H] f16
    const h16_t* __restrict__ WpPack,
    const float* __restrict__ bp,
    float* __restrict__ out)
{
    const int tid  = threadIdx.x;
    const int nt   = blockIdx.x;                       // 0..62
    const int wave = tid >> 5;
    const int lane = tid & 31;
    const int l16  = lane & 15;
    const int hi8  = (lane >> 4) & 1;
    const int brow = wave * 16 + l16;

    v8f acc = {};
    #pragma unroll 1
    for (int kt = 0; kt < HKT; ++kt) {
        v16h a = load_a_f16(hfin + (size_t)brow * Hsz + kt * 32 + hi8 * 8);
        v16h b = *(const v16h*)&WpPack[(((size_t)nt * HKT + kt) * 32 + lane) * 16];
        acc = __builtin_amdgcn_wmma_f32_16x16x32_f16(
            false, a, false, b, (short)0, acc, false, false);
    }
    const int n = nt * 16 + l16;
    if (n < Csz) {
        float bias = bp[n];
        #pragma unroll
        for (int j = 0; j < 8; ++j) {
            int b = wave * 16 + j + hi8 * 8;
            out[(size_t)b * Csz + n] = acc[j] + bias;
        }
    }
}

// ---------------------------------------------------------------------------
extern "C" void kernel_launch(void* const* d_in, const int* in_sizes, int n_in,
                              void* d_out, int out_size, void* d_ws, size_t ws_size,
                              hipStream_t stream) {
    (void)in_sizes; (void)n_in; (void)out_size; (void)ws_size;
    const float* x   = (const float*)d_in[0];
    const float* Wgx = (const float*)d_in[1];
    const float* Wgh = (const float*)d_in[2];
    const float* bg  = (const float*)d_in[3];
    const float* Wix = (const float*)d_in[4];
    const float* Wih = (const float*)d_in[5];
    const float* bi  = (const float*)d_in[6];
    const float* Wfx = (const float*)d_in[7];
    const float* Wfh = (const float*)d_in[8];
    const float* bf  = (const float*)d_in[9];
    const float* Wox = (const float*)d_in[10];
    const float* Woh = (const float*)d_in[11];
    const float* bo  = (const float*)d_in[12];
    const float* Wph = (const float*)d_in[13];
    const float* bp  = (const float*)d_in[14];

    char* ws = (char*)d_ws;
    unsigned int* bar = (unsigned int*)(ws);
    h16_t* Wpack  = (h16_t*)(ws + WS_WPACK_OFF);
    h16_t* WpPack = (h16_t*)(ws + WS_WP_OFF);
    h16_t* hbuf   = (h16_t*)(ws + WS_HBUF_OFF);

    init_state<<<(Bsz * Hsz + 255) / 256, 256, 0, stream>>>(hbuf, bar);
    pack_gates<<<(128 * NKT * 32 + 255) / 256, 256, 0, stream>>>(
        Wgx, Wgh, Wix, Wih, Wfx, Wfh, Wox, Woh, Wpack);
    pack_wp<<<(NTW * HKT * 32 + 255) / 256, 256, 0, stream>>>(Wph, WpPack);

    // 8 col-tiles x 24 K-tiles x 32 lanes x 16 halves x 2B = 196608B LDS
    const size_t ldsBytes = (size_t)8 * NKT * 32 * 16 * sizeof(h16_t);
    lstm_persist<<<NWG, THREADS, ldsBytes, stream>>>(
        x, bg, bi, bf, bo, Wpack, hbuf, bar);

    head_gemm<<<NTW, THREADS, 0, stream>>>(hbuf, WpPack, bp, (float*)d_out);
}
